// Decoder_5669356832200
// MI455X (gfx1250) — matrix operand order repaired and checked by reading, not compiled, not run on hardware
//
#include <hip/hip_runtime.h>
#include <math.h>

// ---------------- CDNA5 WMMA types ----------------
typedef __attribute__((ext_vector_type(16))) __bf16 v16bf;
typedef __attribute__((ext_vector_type(8)))  float  v8f;

union ABf16 {            // 16 bf16 = 32 bytes = two uint4 loads
    v16bf bf;
    unsigned short u[16];
    uint4 q[2];
};
union Acc8 {
    v8f v;
    float f[8];
};

__device__ __forceinline__ unsigned short f2bf(float x) {
    union { float f; unsigned u; } v; v.f = x;
    unsigned r = v.u + 0x7FFFu + ((v.u >> 16) & 1u);   // round-to-nearest-even
    return (unsigned short)(r >> 16);
}

// ---------------- problem constants ----------------
#define Bq   32
#define Hd   512
#define Tq   1500
#define Lq   200
#define NV   999         // V-1 classes
#define NVP  1024        // padded to 64 N-tiles
#define G3   1536        // 3*H
#define KPAD 15          // conv kernel
#define PADC 7
#define TBLK 128         // t-tile per workgroup in scores kernel
#define CWSZ (32 * 32 * 16)   // packed conv_w elements (bf16)

// ---------------- TDM (Tensor Data Mover) support, guarded ----------------
#if defined(__HIP_DEVICE_COMPILE__) && __has_builtin(__builtin_amdgcn_tensor_load_to_lds) && __has_builtin(__builtin_amdgcn_s_wait_tensorcnt)
#define HAVE_TDM 1
#else
#define HAVE_TDM 0
#endif

#if HAVE_TDM
typedef unsigned int u32x4 __attribute__((ext_vector_type(4)));
typedef int          i32x4 __attribute__((ext_vector_type(4)));
typedef int          i32x8 __attribute__((ext_vector_type(8)));

// 1-row 2-byte-element tensor copy: global -> LDS via TDM (D# per ISA ch.8)
__device__ __forceinline__ void tdm_load_2b(const void* g, void* lds_dst, unsigned nelem) {
    unsigned long long ga = (unsigned long long)(uintptr_t)g;
    unsigned la = (unsigned)(uintptr_t)lds_dst;           // LDS aperture: addr[31:0] = offset
    u32x4 g0;
    g0[0] = 1u;                                           // count=1 (valid user descriptor)
    g0[1] = la;                                           // lds_addr
    g0[2] = (unsigned)ga;                                 // global_addr[31:0]
    g0[3] = (unsigned)((ga >> 32) & 0x01FFFFFFu) | (2u << 30);  // addr[56:32] | type=2
    i32x8 g1;
    g1[0] = 0x00010000;                                   // wgmask=0, data_size=1 (2 bytes)
    g1[1] = (int)((nelem & 0xFFFFu) << 16);               // tensor_dim0[15:0]
    g1[2] = (int)(((nelem >> 16) & 0xFFFFu) | (1u << 16));// tensor_dim0[31:16] | tensor_dim1=1
    g1[3] = (int)((nelem & 0xFFFFu) << 16);               // tile_dim0 = nelem
    g1[4] = 1;                                            // tile_dim1=1, tile_dim2=0
    g1[5] = (int)nelem;                                   // tensor_dim0_stride lo
    g1[6] = 0;                                            // stride hi, tensor_dim1_stride lo
    g1[7] = 0;
    i32x4 z = {0, 0, 0, 0};
#if __clang_major__ >= 23
    i32x8 z8 = {0, 0, 0, 0, 0, 0, 0, 0};
    __builtin_amdgcn_tensor_load_to_lds(g0, g1, z, z, z8, 0);
#else
    __builtin_amdgcn_tensor_load_to_lds(g0, g1, z, z, 0);
#endif
}
#endif

// =====================================================================
// weight packing (f32 -> bf16), run once per launch
// =====================================================================
__global__ void pack_bf16(const float* __restrict__ src, unsigned short* __restrict__ dst, int n) {
    int i = blockIdx.x * blockDim.x + threadIdx.x;
    if (i < n) dst[i] = f2bf(src[i]);
}

__global__ void pack_fcw(const float* __restrict__ fcw, unsigned short* __restrict__ dst) {
    int i = blockIdx.x * blockDim.x + threadIdx.x;   // over NVP*Hd
    if (i >= NVP * Hd) return;
    int n = i / Hd, k = i % Hd;
    dst[i] = (n < NV) ? f2bf(fcw[n * Hd + k]) : (unsigned short)0;
}

// conv_w (512,1,15) -> WMMA A-matrix per-lane layout, 32 M-tiles of 16x32 (K zero-padded)
// element j of lane: K = (j<8) ? 8*half+j : 16+8*half+(j-8);  M = mtile*16 + (lane&15)
__global__ void pack_convw(const float* __restrict__ cw, unsigned short* __restrict__ dst) {
    int i = blockIdx.x * blockDim.x + threadIdx.x;   // over CWSZ
    if (i >= CWSZ) return;
    int j    = i & 15;
    int lane = (i >> 4) & 31;
    int mt   = i >> 9;
    int half = lane >> 4;
    int k = (j < 8) ? (8 * half + j) : (16 + 8 * half + (j - 8));
    int h = mt * 16 + (lane & 15);
    float v = (k < KPAD) ? cw[h * KPAD + k] : 0.0f;
    dst[i] = f2bf(v);
}

__global__ void zero_out(float* o) { if (threadIdx.x == 0 && blockIdx.x == 0) o[0] = 0.0f; }

// =====================================================================
// gi = X @ WihT + bih ; gh = Hprev @ WhhT + bhh   (WMMA bf16, M=32)
// X = embedding[tok] (+ sx) built in LDS; Hprev -> bf16 in LDS.
// grid.x = 96 N-tiles, block = 128 (4 waves: {mat,mtile})
// =====================================================================
__global__ void __launch_bounds__(128)
gemm_gates(const float* __restrict__ emb, const int* __restrict__ target, int step,
           const float* __restrict__ sx, int has_sx, const float* __restrict__ hprev,
           const unsigned short* __restrict__ wih, const unsigned short* __restrict__ whh,
           const float* __restrict__ bih, const float* __restrict__ bhh,
           float* __restrict__ gi, float* __restrict__ gh) {
    __shared__ unsigned short xs[Bq * Hd];
    __shared__ unsigned short hs[Bq * Hd];
    int tid = threadIdx.x;

    for (int i = tid * 4; i < Bq * Hd; i += 128 * 4) {
        int b = i >> 9, h = i & (Hd - 1);
        int tok = target[b * Lq + step];
        float4 e = *(const float4*)(emb + tok * Hd + h);
        if (has_sx) {
            float4 s4 = *(const float4*)(sx + i);
            e.x += s4.x; e.y += s4.y; e.z += s4.z; e.w += s4.w;
        }
        xs[i] = f2bf(e.x); xs[i + 1] = f2bf(e.y); xs[i + 2] = f2bf(e.z); xs[i + 3] = f2bf(e.w);
        float4 hv = *(const float4*)(hprev + i);
        hs[i] = f2bf(hv.x); hs[i + 1] = f2bf(hv.y); hs[i + 2] = f2bf(hv.z); hs[i + 3] = f2bf(hv.w);
    }
    __syncthreads();

    int lane  = tid & 31;
    int w     = tid >> 5;
    int mtile = w & 1;
    int mat   = w >> 1;
    int ntile = blockIdx.x;

    const unsigned short* A  = mat ? hs : xs;
    const unsigned short* Bm = mat ? whh : wih;
    const float* bias = mat ? bhh : bih;
    float* out = mat ? gh : gi;

    int nloc = lane & 15, half = lane >> 4;
    int m  = mtile * 16 + nloc;          // batch row for A
    int nc = ntile * 16 + nloc;          // output column (row of W)

    Acc8 c;
#pragma unroll
    for (int r = 0; r < 8; ++r) c.f[r] = 0.0f;

#pragma unroll 4
    for (int kt = 0; kt < 16; ++kt) {
        int k0 = kt * 32;
        ABf16 a, bm;
        a.q[0] = *(const uint4*)(A + m * Hd + k0 + 16 * half);
        a.q[1] = *(const uint4*)(A + m * Hd + k0 + 16 * half + 8);
        const uint4* pb = (const uint4*)(Bm + nc * Hd + k0 + 16 * half);
        bm.q[0] = pb[0]; bm.q[1] = pb[1];
        c.v = __builtin_amdgcn_wmma_f32_16x16x32_bf16(false, a.bf, false, bm.bf,
                                                      (short)0, c.v, false, false);
    }
#pragma unroll
    for (int r = 0; r < 8; ++r) {
        int mm = mtile * 16 + 8 * half + r;
        out[mm * G3 + nc] = c.f[r] + bias[nc];
    }
}

// =====================================================================
// GRU gate nonlinearity -> hidden
// =====================================================================
__global__ void gru_gates(const float* __restrict__ gi, const float* __restrict__ gh,
                          const float* __restrict__ hprev, float* __restrict__ hidden) {
    int i = blockIdx.x * blockDim.x + threadIdx.x;   // B*H
    if (i >= Bq * Hd) return;
    int b = i >> 9, h = i & (Hd - 1);
    int base = b * G3;
    float r = 1.0f / (1.0f + __expf(-(gi[base + h] + gh[base + h])));
    float z = 1.0f / (1.0f + __expf(-(gi[base + Hd + h] + gh[base + Hd + h])));
    float n = tanhf(gi[base + 2 * Hd + h] + r * gh[base + 2 * Hd + h]);
    hidden[i] = (1.0f - z) * n + z * hprev[i];
}

// =====================================================================
// Fused attention scores:
//   cterm(h,t) = conv(ax) via WMMA (A = TDM-staged conv_w tiles, B = ax windows)
//   scores[b,t] = attn_b + sum_h attn_w[h]*relu(enc[b,t,h]+hid[b,h]+conv_b[h]+cterm)
// grid = (ceil(T/TBLK), B), block = 256 (8 waves x 16 t each)
// =====================================================================
__global__ void __launch_bounds__(256)
attn_scores(const float* __restrict__ enc, const float* __restrict__ hidden,
            const float* __restrict__ ax,
            const unsigned short* __restrict__ convw_packed,
            const float* __restrict__ conv_b, const float* __restrict__ attn_w,
            const float* __restrict__ attn_b, int has_ax,
            float* __restrict__ scores) {
    __shared__ unsigned short cw_s[CWSZ];   // 32 KB of WMMA A-tiles
    __shared__ float base_s[Hd];
    __shared__ float aw_s[Hd];
    __shared__ float axwin[TBLK + 32];

    int tid = threadIdx.x;
    int b = blockIdx.y;
    int t0 = blockIdx.x * TBLK;

    if (has_ax) {
#if HAVE_TDM
        if (tid == 0) {
            tdm_load_2b(convw_packed, cw_s, CWSZ);     // async DMA into LDS
            __builtin_amdgcn_s_wait_tensorcnt(0);
        }
#else
        for (int i = tid; i < CWSZ / 8; i += 256)
            ((uint4*)cw_s)[i] = ((const uint4*)convw_packed)[i];
#endif
        for (int i = tid; i < TBLK + 32; i += 256) {
            int t = t0 + i - 16;
            axwin[i] = (t >= 0 && t < Tq) ? ax[b * Tq + t] : 0.0f;
        }
    }
    for (int i = tid; i < Hd; i += 256) {
        base_s[i] = hidden[b * Hd + i] + (has_ax ? conv_b[i] : 0.0f);
        aw_s[i] = attn_w[i];
    }
    __syncthreads();

    int lane = tid & 31, wave = tid >> 5;
    int nloc = lane & 15, half = lane >> 4;
    int tsub = t0 + wave * 16;
    int t = tsub + nloc;

    // B-matrix: B[k][n] = ax[b, tsub+n + k - PADC], K zero-padded past 15
    ABf16 bm;
#pragma unroll
    for (int j = 0; j < 16; ++j) {
        int k = (j < 8) ? (8 * half + j) : (16 + 8 * half + (j - 8));
        float v = 0.0f;
        if (has_ax && k < KPAD)
            v = axwin[(wave * 16) + nloc + k - PADC + 16];
        bm.u[j] = f2bf(v);
    }

    float partial = 0.0f;
    for (int mt = 0; mt < 32; ++mt) {
        Acc8 c;
#pragma unroll
        for (int r = 0; r < 8; ++r) c.f[r] = 0.0f;
        if (has_ax) {
            ABf16 a;
            const uint4* pa = (const uint4*)(cw_s + ((mt * 32) + lane) * 16);
            a.q[0] = pa[0]; a.q[1] = pa[1];
            c.v = __builtin_amdgcn_wmma_f32_16x16x32_bf16(false, a.bf, false, bm.bf,
                                                          (short)0, c.v, false, false);
        }
        if (t < Tq) {
            const float* ep = enc + ((size_t)(b * Tq + t)) * Hd + mt * 16 + 8 * half;
            __builtin_prefetch(ep + 16, 0, 0);   // next h-tile (L2-resident stream)
            float4 e0 = *(const float4*)ep;
            float4 e1 = *(const float4*)(ep + 4);
            float ev[8] = {e0.x, e0.y, e0.z, e0.w, e1.x, e1.y, e1.z, e1.w};
#pragma unroll
            for (int r = 0; r < 8; ++r) {
                int h = mt * 16 + 8 * half + r;
                float p = ev[r] + base_s[h] + c.f[r];
                p = fmaxf(p, 0.0f);
                partial += aw_s[h] * p;
            }
        }
    }
    partial += __shfl_xor(partial, 16, 32);
    if (half == 0 && t < Tq) scores[b * Tq + t] = partial + attn_b[0];
}

// =====================================================================
// ax = softmax(log_t * scores) along T, then sx = enc^T ax (fused)
// one block (512 threads) per batch row
// =====================================================================
__global__ void __launch_bounds__(512)
softmax_context(const float* __restrict__ scores, float scale,
                const float* __restrict__ enc,
                float* __restrict__ ax, float* __restrict__ sx) {
    __shared__ float red[512];
    __shared__ float ax_s[Tq];
    int b = blockIdx.x, tid = threadIdx.x;
    const float* row = scores + b * Tq;

    float m = -3.4e38f;
    for (int t = tid; t < Tq; t += 512) m = fmaxf(m, scale * row[t]);
    red[tid] = m; __syncthreads();
    for (int s = 256; s > 0; s >>= 1) {
        if (tid < s) red[tid] = fmaxf(red[tid], red[tid + s]);
        __syncthreads();
    }
    m = red[0];
    __syncthreads();

    float sum = 0.0f;
    for (int t = tid; t < Tq; t += 512) sum += __expf(scale * row[t] - m);
    red[tid] = sum; __syncthreads();
    for (int s = 256; s > 0; s >>= 1) {
        if (tid < s) red[tid] += red[tid + s];
        __syncthreads();
    }
    float inv = 1.0f / red[0];

    for (int t = tid; t < Tq; t += 512) {
        float a = __expf(scale * row[t] - m) * inv;
        ax_s[t] = a;
        ax[b * Tq + t] = a;
    }
    __syncthreads();

    const float* ep = enc + (size_t)b * Tq * Hd + tid;
    float acc = 0.0f;
#pragma unroll 4
    for (int t = 0; t < Tq; ++t) {
        __builtin_prefetch(ep + (t + 8) * Hd, 0, 0);
        acc += ep[t * Hd] * ax_s[t];
    }
    sx[b * Hd + tid] = acc;
}

// =====================================================================
// logits = (hidden+sx) @ fcwT + fcb  (WMMA bf16; Y staged bf16 in LDS)
// grid.x = 32 (2 N-tiles each), block = 128 (4 waves: {ntile,mtile})
// =====================================================================
__global__ void __launch_bounds__(128)
gemm_logits(const float* __restrict__ hidden, const float* __restrict__ sx,
            const unsigned short* __restrict__ fcw, const float* __restrict__ fcb,
            float* __restrict__ logits) {
    __shared__ unsigned short ys[Bq * Hd];
    int tid = threadIdx.x;
    for (int i = tid * 4; i < Bq * Hd; i += 128 * 4) {
        float4 hv = *(const float4*)(hidden + i);
        float4 s4 = *(const float4*)(sx + i);
        ys[i] = f2bf(hv.x + s4.x); ys[i + 1] = f2bf(hv.y + s4.y);
        ys[i + 2] = f2bf(hv.z + s4.z); ys[i + 3] = f2bf(hv.w + s4.w);
    }
    __syncthreads();

    int lane = tid & 31;
    int w = tid >> 5;
    int mtile = w & 1;
    int ntile = blockIdx.x * 2 + (w >> 1);
    int nloc = lane & 15, half = lane >> 4;
    int m = mtile * 16 + nloc;
    int nc = ntile * 16 + nloc;

    Acc8 c;
#pragma unroll
    for (int r = 0; r < 8; ++r) c.f[r] = 0.0f;

#pragma unroll 4
    for (int kt = 0; kt < 16; ++kt) {
        int k0 = kt * 32;
        ABf16 a, bm;
        a.q[0] = *(const uint4*)(ys + m * Hd + k0 + 16 * half);
        a.q[1] = *(const uint4*)(ys + m * Hd + k0 + 16 * half + 8);
        const uint4* pb = (const uint4*)(fcw + nc * Hd + k0 + 16 * half);
        bm.q[0] = pb[0]; bm.q[1] = pb[1];
        c.v = __builtin_amdgcn_wmma_f32_16x16x32_bf16(false, a.bf, false, bm.bf,
                                                      (short)0, c.v, false, false);
    }
#pragma unroll
    for (int r = 0; r < 8; ++r) {
        int mm = mtile * 16 + 8 * half + r;
        if (nc < NV) logits[mm * NVP + nc] = c.f[r] + fcb[nc];
    }
}

// =====================================================================
// loss += logsumexp(logits[b]) - logits[b, label]   (one block per b)
// =====================================================================
__global__ void ce_reduce(const float* __restrict__ logits, const int* __restrict__ target,
                          int step, float* __restrict__ out) {
    __shared__ float red[256];
    __shared__ float s_max;
    int b = blockIdx.x, tid = threadIdx.x;
    const float* row = logits + b * NVP;

    float m = -3.4e38f;
    for (int i = tid; i < NV; i += 256) m = fmaxf(m, row[i]);
    red[tid] = m; __syncthreads();
    for (int s = 128; s > 0; s >>= 1) {
        if (tid < s) red[tid] = fmaxf(red[tid], red[tid + s]);
        __syncthreads();
    }
    if (tid == 0) s_max = red[0];
    __syncthreads();

    float sum = 0.0f;
    for (int i = tid; i < NV; i += 256) sum += __expf(row[i] - s_max);
    red[tid] = sum; __syncthreads();
    for (int s = 128; s > 0; s >>= 1) {
        if (tid < s) red[tid] += red[tid + s];
        __syncthreads();
    }
    if (tid == 0) {
        int lbl = target[b * Lq + step + 1];
        float contrib = __logf(red[0]) + s_max - row[lbl];
        atomicAdd(out, contrib);
    }
}

// =====================================================================
// host
// =====================================================================
static inline size_t bump(size_t& o, size_t bytes) {
    size_t r = o;
    o = (o + bytes + 255) & ~(size_t)255;
    return r;
}

extern "C" void kernel_launch(void* const* d_in, const int* in_sizes, int n_in,
                              void* d_out, int out_size, void* d_ws, size_t ws_size,
                              hipStream_t stream) {
    const int*   target  = (const int*)  d_in[0];
    const float* enc_out = (const float*)d_in[1];
    const float* enc_hid = (const float*)d_in[2];
    const float* emb     = (const float*)d_in[3];
    const float* conv_w  = (const float*)d_in[4];
    const float* conv_b  = (const float*)d_in[5];
    const float* attn_w  = (const float*)d_in[6];
    const float* attn_b  = (const float*)d_in[7];
    const float* w_ih    = (const float*)d_in[8];
    const float* b_ih    = (const float*)d_in[9];
    const float* w_hh    = (const float*)d_in[10];
    const float* b_hh    = (const float*)d_in[11];
    const float* fc_w    = (const float*)d_in[12];
    const float* fc_b    = (const float*)d_in[13];
    float* out = (float*)d_out;

    char* ws = (char*)d_ws;
    size_t o = 0;
    unsigned short* wih_bf = (unsigned short*)(ws + bump(o, (size_t)G3 * Hd * 2));
    unsigned short* whh_bf = (unsigned short*)(ws + bump(o, (size_t)G3 * Hd * 2));
    unsigned short* fcw_bf = (unsigned short*)(ws + bump(o, (size_t)NVP * Hd * 2));
    unsigned short* cw_pk  = (unsigned short*)(ws + bump(o, CWSZ * 2));
    float* gi     = (float*)(ws + bump(o, (size_t)Bq * G3 * 4));
    float* gh     = (float*)(ws + bump(o, (size_t)Bq * G3 * 4));
    float* hidden = (float*)(ws + bump(o, Bq * Hd * 4));
    float* sx     = (float*)(ws + bump(o, Bq * Hd * 4));
    float* scores = (float*)(ws + bump(o, (size_t)Bq * Tq * 4));
    float* ax     = (float*)(ws + bump(o, (size_t)Bq * Tq * 4));
    float* logits = (float*)(ws + bump(o, (size_t)Bq * NVP * 4));
    (void)ws_size; (void)n_in; (void)in_sizes; (void)out_size;

    // one-time packing (re-done each call: deterministic, cheap)
    pack_bf16<<<(G3 * Hd + 255) / 256, 256, 0, stream>>>(w_ih, wih_bf, G3 * Hd);
    pack_bf16<<<(G3 * Hd + 255) / 256, 256, 0, stream>>>(w_hh, whh_bf, G3 * Hd);
    pack_fcw <<<(NVP * Hd + 255) / 256, 256, 0, stream>>>(fc_w, fcw_bf);
    pack_convw<<<(CWSZ + 255) / 256, 256, 0, stream>>>(conv_w, cw_pk);
    zero_out<<<1, 1, 0, stream>>>(out);

    const float scale = logf((float)Tq);
    const int nEl = Bq * Hd;
    const int tblocks = (Tq + TBLK - 1) / TBLK;

    for (int it = 0; it < Lq - 1; ++it) {
        const int has_ax = (it > 0);
        const float* hprev = (it == 0) ? enc_hid : hidden;

        gemm_gates<<<G3 / 16, 128, 0, stream>>>(emb, target, it, sx, has_ax, hprev,
                                                wih_bf, whh_bf, b_ih, b_hh, gi, gh);
        gru_gates<<<(nEl + 255) / 256, 256, 0, stream>>>(gi, gh, hprev, hidden);
        attn_scores<<<dim3(tblocks, Bq), 256, 0, stream>>>(enc_out, hidden, ax, cw_pk,
                                                           conv_b, attn_w, attn_b,
                                                           has_ax, scores);
        softmax_context<<<Bq, 512, 0, stream>>>(scores, scale, enc_out, ax, sx);
        gemm_logits<<<NVP / 32, 128, 0, stream>>>(hidden, sx, fcw_bf, fc_b, logits);
        ce_reduce<<<Bq, 256, 0, stream>>>(logits, target, it, out);
    }
}
